// IAF_PointNetFeaturePropagation_37014028157087
// MI455X (gfx1250) — compile-verified, matrix operand order checked by reading, not run on hardware
//
#include <hip/hip_runtime.h>

// ---------------------------------------------------------------------------
// MI455X / gfx1250 implementation. wave32, WMMA bf16 (16x16x32) for all GEMMs.
// Activations stored feature-last so WMMA A/B fragments are contiguous 32B
// loads; 192MB L2 holds the whole working set so GEMMs stream from global.
// GEMM K-loop is a compile-time fully-unrolled pipeline: fragments for k and
// k+32 are loaded before either wmma group, each unrolled step uses fresh
// virtual registers (no cross-iteration copies -> no WMMA/VALU hazard nops),
// letting the scheduler emit partial s_wait_loadcnt and overlap L2 latency.
// __launch_bounds__(128) on the WMMA kernels declares the real 4-wave blocks
// so the allocator budgets 1 wave/SIMD and grants the full VGPR file
// (default 1024-thread assumption capped waves at 128 VGPRs -> spills).
// ---------------------------------------------------------------------------

typedef __attribute__((ext_vector_type(16))) __bf16 bf16x16;
typedef __attribute__((ext_vector_type(8)))  float  f32x8;

#define WMMA_BF16(a, b, c) \
  __builtin_amdgcn_wmma_f32_16x16x32_bf16(false, (a), false, (b), (short)0, (c), false, false)

constexpr int Bb   = 4;
constexpr int Nn   = 8192;
constexpr int Ss   = 2048;
constexpr int Kk   = 20;
constexpr int CLS  = 21;
constexpr int CO   = 128;
constexpr int Hn   = 2047;   // outer(1638) + inner(409)
constexpr int Hpad = 2048;
constexpr int OUTER = 1638;
constexpr int INNER = 409;

// ---------------------------------------------------------------------------
// Weight fp32 -> bf16 with zero padding (O -> Op rows, C -> Cp cols)
// ---------------------------------------------------------------------------
__global__ void cvt_pad_kernel(const float* __restrict__ src, __bf16* __restrict__ dst,
                               int O, int Op, int C, int Cp) {
  int i = blockIdx.x * 256 + threadIdx.x;
  if (i >= Op * Cp) return;
  int o = i / Cp, c = i % Cp;
  dst[i] = (o < O && c < C) ? (__bf16)src[o * C + c] : (__bf16)0.f;
}

__global__ void pad_bias_kernel(const float* __restrict__ src, float* __restrict__ dst,
                                int n, int np) {
  int i = threadIdx.x;
  if (i < np) dst[i] = (i < n) ? src[i] : 0.f;
}

// ---------------------------------------------------------------------------
// 3-NN inverse-distance interpolation. Block handles 256 query points of one
// batch; xyz2 of that batch staged in LDS (24KB). Emits bf16 concat input
// X0[b][n][0:256]=points1, [256:512]=interp, and f32 interp_preds.
// ---------------------------------------------------------------------------
__global__ void knn_interp_kernel(const float* __restrict__ xyz1,
                                  const float* __restrict__ xyz2,
                                  const float* __restrict__ points1,
                                  const float* __restrict__ points2,
                                  const float* __restrict__ lastpred,
                                  __bf16* __restrict__ X0,
                                  float* __restrict__ ipreds) {
  __shared__ float sx[Ss], sy[Ss], sz[Ss];
  const int bpb = Nn / 256;
  int b = blockIdx.x / bpb;
  int n = (blockIdx.x % bpb) * 256 + threadIdx.x;
  const float* x2 = xyz2 + (long)b * Ss * 3;
  for (int s = threadIdx.x; s < Ss; s += 256) {
    sx[s] = x2[s * 3 + 0];
    sy[s] = x2[s * 3 + 1];
    sz[s] = x2[s * 3 + 2];
  }
  __syncthreads();
  const float* p = xyz1 + ((long)b * Nn + n) * 3;
  float px = p[0], py = p[1], pz = p[2];
  float d0 = 3e38f, d1 = 3e38f, d2 = 3e38f;
  int i0 = 0, i1 = 0, i2 = 0;
  for (int s = 0; s < Ss; s++) {
    float dx = px - sx[s], dy = py - sy[s], dz = pz - sz[s];
    float d = dx * dx + dy * dy + dz * dz;
    if (d < d0)      { d2 = d1; i2 = i1; d1 = d0; i1 = i0; d0 = d; i0 = s; }
    else if (d < d1) { d2 = d1; i2 = i1; d1 = d;  i1 = s; }
    else if (d < d2) { d2 = d;  i2 = s; }
  }
  float w0 = 1.f / fmaxf(d0, 1e-10f);
  float w1 = 1.f / fmaxf(d1, 1e-10f);
  float w2 = 1.f / fmaxf(d2, 1e-10f);
  float wsum = w0 + w1 + w2;
  w0 /= wsum; w1 /= wsum; w2 /= wsum;

  __bf16* xrow = X0 + ((long)b * Nn + n) * 512;
  const float* p1b = points1 + (long)b * 256 * Nn;
  const float* p2b = points2 + (long)b * 256 * Ss;
  for (int c = 0; c < 256; c++) {
    xrow[c] = (__bf16)p1b[(long)c * Nn + n];
    float iv = w0 * p2b[(long)c * Ss + i0] + w1 * p2b[(long)c * Ss + i1] +
               w2 * p2b[(long)c * Ss + i2];
    xrow[256 + c] = (__bf16)iv;
  }
  const float* lpb = lastpred + (long)b * Ss * CLS;
  float* iprow = ipreds + ((long)b * Nn + n) * CLS;
  for (int c = 0; c < CLS; c++)
    iprow[c] = w0 * lpb[(long)i0 * CLS + c] + w1 * lpb[(long)i1 * CLS + c] +
               w2 * lpb[(long)i2 * CLS + c];
}

// ---------------------------------------------------------------------------
// Generic bf16 WMMA GEMM: Y[n][o] = act( (sum_c W[o][c]*X[n][c] + bias[o])
//                                        * g[o]/sqrt(1+eps) + beta[o] + res[n][o] )
// KDIM is compile-time (multiple of 32). Wave computes 32(o) x 64(n): 8 wmma
// per 32-k step, steps paired and fully unrolled. Block = 4 waves covering
// 128 consecutive o. Odim must be a multiple of 32.
// Dual store: bf16 (optionally [o][n]) and f32 (optionally [o][n]).
// ---------------------------------------------------------------------------
template <int KDIM>
__global__ void __launch_bounds__(128)
gemm_bf16(const __bf16* __restrict__ W, int ldw,
          const __bf16* __restrict__ X, long strideX, int ldx,
          int Nvalid, int Odim,
          const float* __restrict__ bias,
          const float* __restrict__ gamma,
          const float* __restrict__ beta,
          const float* __restrict__ residual, long strideRes, int ldres,
          int actmode,  // 0 none, 1 relu, 2 leaky(0.2)
          __bf16* __restrict__ outBf, long strideOB, int ldob, int obTrans,
          float* __restrict__ outF, long strideOF, int ldof, int ofTrans) {
  int lane = threadIdx.x & 31;
  int wv   = threadIdx.x >> 5;
  int l16  = lane & 15;
  int half = lane >> 4;
  int o0 = blockIdx.y * 128 + wv * 32;
  if (o0 >= Odim) return;
  int n0 = blockIdx.x * 64;
  int bz = blockIdx.z;

  const __bf16* Xb   = X + (long)bz * strideX;
  const __bf16* aptr = W  + (long)(o0 + l16) * ldw + half * 16;   // A: row l16, K-half
  const __bf16* bptr = Xb + (long)(n0 + l16) * ldx + half * 16;   // B: col l16, K-half

  f32x8 acc[2][4] = {};

  auto ldfrags = [&](int k0, bf16x16 (&fr)[6]) {
    fr[0] = *(const bf16x16*)(aptr + k0);
    fr[1] = *(const bf16x16*)(aptr + (long)16 * ldw + k0);
#pragma unroll
    for (int t = 0; t < 4; t++)
      fr[2 + t] = *(const bf16x16*)(bptr + (long)(16 * t) * ldx + k0);
  };
  auto mm = [&](bf16x16 (&fr)[6]) {
#pragma unroll
    for (int i = 0; i < 2; i++)
#pragma unroll
      for (int t = 0; t < 4; t++)
        acc[i][t] = WMMA_BF16(fr[i], fr[2 + t], acc[i][t]);
  };

#pragma unroll
  for (int k0 = 0; k0 + 64 <= KDIM; k0 += 64) {
    bf16x16 f0[6], f1[6];
    ldfrags(k0, f0);        // both fragment sets issued before either wmma
    ldfrags(k0 + 32, f1);   // group -> partial loadcnt waits, overlap
    mm(f0);
    mm(f1);
  }
  if constexpr ((KDIM & 63) != 0) {
    bf16x16 f0[6];
    ldfrags(KDIM - 32, f0);
    mm(f0);
  }

  const float invs = rsqrtf(1.0f + 1e-5f);
#pragma unroll
  for (int i = 0; i < 2; i++) {
    for (int v = 0; v < 8; v++) {
      int o = o0 + i * 16 + v + 8 * half;             // D: M = v + 8*(lane/16)
      float bsv = bias  ? bias[o]         : 0.f;
      float sc  = gamma ? gamma[o] * invs : 1.f;
      float sh  = beta  ? beta[o]         : 0.f;
      for (int t = 0; t < 4; t++) {
        int n = n0 + t * 16 + l16;                    // D: N = lane%16
        if (n >= Nvalid) continue;
        float y = acc[i][t][v] + bsv;
        y = y * sc + sh;
        if (residual) y += residual[(long)bz * strideRes + (long)n * ldres + o];
        if (actmode == 1)      y = fmaxf(y, 0.f);
        else if (actmode == 2) y = (y >= 0.f) ? y : 0.2f * y;
        if (outBf) {
          long idx = obTrans ? ((long)o * ldob + n) : ((long)n * ldob + o);
          outBf[(long)bz * strideOB + idx] = (__bf16)y;
        }
        if (outF) {
          long idx = ofTrans ? ((long)o * ldof + n) : ((long)n * ldof + o);
          outF[(long)bz * strideOF + idx] = y;
        }
      }
    }
  }
}

// ---------------------------------------------------------------------------
// diff[b][n] = sum_{k,c} (xt[idx1[b,n,k]][c] - xt[b,n][c])^2 ; wave per point.
// ---------------------------------------------------------------------------
__global__ void diff_kernel(const float* __restrict__ xt, const int* __restrict__ idx1,
                            float* __restrict__ diff) {
  int lane = threadIdx.x & 31, wv = threadIdx.x >> 5;
  long gid = (long)blockIdx.x * 4 + wv;        // 0 .. B*N-1
  const float* xr = xt + gid * CO + lane * 4;
  float x0 = xr[0], x1 = xr[1], x2 = xr[2], x3 = xr[3];
  const int* ix = idx1 + gid * Kk;
  float acc = 0.f;
  for (int k = 0; k < Kk; k++) {
    const float* fr = xt + (long)ix[k] * CO + lane * 4;
    float a = fr[0] - x0, b = fr[1] - x1, c = fr[2] - x2, d = fr[3] - x3;
    acc += a * a + b * b + c * c + d * d;
  }
  for (int off = 1; off < 32; off <<= 1) acc += __shfl_xor(acc, off, 32);
  if (lane == 0) diff[gid] = acc;
}

// ---------------------------------------------------------------------------
// Per-batch bitonic sort (descending value, ascending index tie-break) of the
// 8192 diffs in 64KB LDS; emits H_idx = top-OUTER indices ++ INNER pseudo-
// random distinct indices (odd-multiplier bijection mod 8192).
// min/max normalization is monotone so top-k on raw diff is identical.
// ---------------------------------------------------------------------------
__global__ void topk_kernel(const float* __restrict__ diff, int* __restrict__ Hidx) {
  __shared__ float sv[Nn];
  __shared__ int   si[Nn];
  int b = blockIdx.x, tid = threadIdx.x, bs = blockDim.x;
  for (int i = tid; i < Nn; i += bs) { sv[i] = diff[(long)b * Nn + i]; si[i] = i; }
  __syncthreads();
  for (int k = 2; k <= Nn; k <<= 1) {
    for (int j = k >> 1; j > 0; j >>= 1) {
      for (int t = tid; t < Nn; t += bs) {
        int ixj = t ^ j;
        if (ixj > t) {
          float vi = sv[t], vj = sv[ixj];
          int   ii = si[t], ij = si[ixj];
          bool desc = ((t & k) == 0);
          bool jFirst = (vj > vi) || (vj == vi && ij < ii);
          bool iFirst = (vi > vj) || (vi == vj && ii < ij);
          if (desc ? jFirst : iFirst) {
            sv[t] = vj; sv[ixj] = vi; si[t] = ij; si[ixj] = ii;
          }
        }
      }
      __syncthreads();
    }
  }
  for (int t = tid; t < Hpad; t += bs) {
    int v;
    if (t < OUTER)              v = si[t];
    else if (t < OUTER + INNER) v = (int)(((unsigned)(t - OUTER) * 2654435761u) & (Nn - 1));
    else                        v = 0;
    Hidx[b * Hpad + t] = v;
  }
}

// ---------------------------------------------------------------------------
// XH[b][h][0:128]=xt[H_idx], [128:149]=interp_preds[H_idx], [149:160]=0,
// padded row 2047 zeroed (keeps downstream GEMMs deterministic).
// ---------------------------------------------------------------------------
__global__ void build_xh_kernel(const float* __restrict__ xt,
                                const float* __restrict__ ipreds,
                                const int* __restrict__ Hidx,
                                __bf16* __restrict__ XH) {
  int b = blockIdx.y, h = blockIdx.x, c = threadIdx.x;
  if (c >= 160) return;
  __bf16 v = (__bf16)0.f;
  if (h < Hn) {
    int src = Hidx[b * Hpad + h];
    if (c < CO)            v = (__bf16)xt[((long)b * Nn + src) * CO + c];
    else if (c < CO + CLS) v = (__bf16)ipreds[((long)b * Nn + src) * CLS + (c - CO)];
  }
  XH[((long)b * Hpad + h) * 160 + c] = v;
}

// ---------------------------------------------------------------------------
// Flash attention: per wave 16 queries x all 2047 keys, dim 128.
// QK^T and AV both via v_wmma_f32_16x16x32_bf16, fragments loaded in groups
// of 4 (32 VGPRs); with __launch_bounds__(128) the whole wave state (Q
// fragments, 8 f32 accumulators, softmax stats) stays register-resident.
// Online softmax with half-wave __shfl_xor row reductions; P re-laid D->A
// through LDS.
// ---------------------------------------------------------------------------
__global__ void __launch_bounds__(128)
attn_kernel(const __bf16* __restrict__ theta,
            const __bf16* __restrict__ phi,
            const __bf16* __restrict__ gxT,   // [b][c][Hpad]
            __bf16* __restrict__ yout) {
  __shared__ float sP[4][16][32];
  int lane = threadIdx.x & 31, wv = threadIdx.x >> 5;
  int l16 = lane & 15, half = lane >> 4;
  int b = blockIdx.z;
  int q0 = (blockIdx.x * 4 + wv) * 16;

  const __bf16* th = theta + ((long)b * Nn + q0) * CO;
  bf16x16 qf[4];
#pragma unroll
  for (int kk = 0; kk < 4; kk++)
    qf[kk] = *(const bf16x16*)(th + (long)l16 * CO + kk * 32 + half * 16);

  f32x8 oacc[8] = {};
  float m[8], l[8];
  for (int v = 0; v < 8; v++) { m[v] = -3.0e38f; l[v] = 0.f; }

  const __bf16* phb = phi + (long)b * Hpad * CO;
  const __bf16* gxb = gxT + (long)b * CO * Hpad;

  for (int h0 = 0; h0 < Hpad; h0 += 32) {
    // ---- QK^T: two groups of 4 phi fragments + 4 wmmas each ----
    f32x8 f0 = {}, f1 = {};
#pragma unroll
    for (int g = 0; g < 2; g++) {
      const __bf16* pr0 = phb + (long)(h0 + l16) * CO + (2 * g) * 32 + half * 16;
      const __bf16* pr1 = phb + (long)(h0 + 16 + l16) * CO + (2 * g) * 32 + half * 16;
      bf16x16 pk[4];
      pk[0] = *(const bf16x16*)(pr0);
      pk[1] = *(const bf16x16*)(pr1);
      pk[2] = *(const bf16x16*)(pr0 + 32);
      pk[3] = *(const bf16x16*)(pr1 + 32);
      f0 = WMMA_BF16(qf[2 * g],     pk[0], f0);
      f1 = WMMA_BF16(qf[2 * g],     pk[1], f1);
      f0 = WMMA_BF16(qf[2 * g + 1], pk[2], f0);
      f1 = WMMA_BF16(qf[2 * g + 1], pk[3], f1);
    }
    bool msk0 = (h0 + l16) >= Hn;
    bool msk1 = (h0 + 16 + l16) >= Hn;

    __syncthreads();  // protect previous iteration's sP reads
    for (int v = 0; v < 8; v++) {
      float a0 = msk0 ? -3.0e38f : f0[v];
      float a1 = msk1 ? -3.0e38f : f1[v];
      float rm = fmaxf(a0, a1);
      for (int off = 1; off < 16; off <<= 1) rm = fmaxf(rm, __shfl_xor(rm, off, 32));
      float mn   = fmaxf(m[v], rm);
      float corr = __expf(m[v] - mn);
      m[v] = mn;
      float p0 = msk0 ? 0.f : __expf(a0 - mn);
      float p1 = msk1 ? 0.f : __expf(a1 - mn);
      float rs = p0 + p1;
      for (int off = 1; off < 16; off <<= 1) rs += __shfl_xor(rs, off, 32);
      l[v] = l[v] * corr + rs;
      for (int t = 0; t < 8; t++) oacc[t][v] = oacc[t][v] * corr;
      int row = v + 8 * half;                 // D layout row
      sP[wv][row][l16]      = p0;
      sP[wv][row][16 + l16] = p1;
    }
    __syncthreads();

    bf16x16 pf;                                // A-frag: row l16, K-half
    {
      const float* rp = &sP[wv][l16][half * 16];
#pragma unroll
      for (int j = 0; j < 16; j++) pf[j] = (__bf16)rp[j];
    }
    // ---- AV: two groups of 4 g_x fragments + 4 wmmas each ----
#pragma unroll
    for (int g = 0; g < 2; g++) {
      bf16x16 gb[4];
#pragma unroll
      for (int t = 0; t < 4; t++)
        gb[t] = *(const bf16x16*)(gxb + (long)((4 * g + t) * 16 + l16) * Hpad + h0 + half * 16);
#pragma unroll
      for (int t = 0; t < 4; t++)
        oacc[4 * g + t] = WMMA_BF16(pf, gb[t], oacc[4 * g + t]);
    }
  }

#pragma unroll
  for (int t = 0; t < 8; t++)
    for (int v = 0; v < 8; v++) {
      int q = q0 + v + 8 * half;
      float val = oacc[t][v] / l[v];
      yout[((long)b * Nn + q) * CO + t * 16 + l16] = (__bf16)val;
    }
}

// ---------------------------------------------------------------------------
// log_softmax over 21 classes; logits stored [n][32] (rows 21..31 padded).
// ---------------------------------------------------------------------------
__global__ void logsoftmax_kernel(const float* __restrict__ logits,
                                  float* __restrict__ out) {
  long gid = (long)blockIdx.x * 256 + threadIdx.x;   // 0 .. B*N-1
  const float* lr = logits + gid * 32;
  float mx = -3e38f;
  for (int c = 0; c < CLS; c++) mx = fmaxf(mx, lr[c]);
  float s = 0.f;
  for (int c = 0; c < CLS; c++) s += __expf(lr[c] - mx);
  float lse = mx + __logf(s);
  float* orow = out + gid * CLS;
  for (int c = 0; c < CLS; c++) orow[c] = lr[c] - lse;
}

// ---------------------------------------------------------------------------
extern "C" void kernel_launch(void* const* d_in, const int* in_sizes, int n_in,
                              void* d_out, int out_size, void* d_ws, size_t ws_size,
                              hipStream_t stream) {
  const float* xyz1    = (const float*)d_in[0];
  const float* xyz2    = (const float*)d_in[1];
  const float* points1 = (const float*)d_in[2];
  const float* points2 = (const float*)d_in[3];
  const float* lastpred= (const float*)d_in[5];
  const float* w_mlp0  = (const float*)d_in[6];
  const float* b_mlp0  = (const float*)d_in[7];
  const float* g_bn0   = (const float*)d_in[8];
  const float* be_bn0  = (const float*)d_in[9];
  const float* w_mlp1  = (const float*)d_in[10];
  const float* b_mlp1  = (const float*)d_in[11];
  const float* g_bn1   = (const float*)d_in[12];
  const float* be_bn1  = (const float*)d_in[13];
  const float* w_h     = (const float*)d_in[14];
  const float* g_hbn   = (const float*)d_in[15];
  const float* be_hbn  = (const float*)d_in[16];
  const float* w_outm  = (const float*)d_in[17];
  const float* g_obn   = (const float*)d_in[18];
  const float* be_obn  = (const float*)d_in[19];
  const float* w_g     = (const float*)d_in[20];
  const float* b_g     = (const float*)d_in[21];
  const float* w_theta = (const float*)d_in[22];
  const float* b_theta = (const float*)d_in[23];
  const float* w_phi   = (const float*)d_in[24];
  const float* b_phi   = (const float*)d_in[25];
  const float* w_W     = (const float*)d_in[26];
  const float* b_W     = (const float*)d_in[27];
  const float* g_Wbn   = (const float*)d_in[28];
  const float* be_Wbn  = (const float*)d_in[29];
  const float* w5      = (const float*)d_in[30];
  const float* b5      = (const float*)d_in[31];
  const float* g_bn5   = (const float*)d_in[32];
  const float* be_bn5  = (const float*)d_in[33];
  const float* w6      = (const float*)d_in[34];
  const float* b6      = (const float*)d_in[35];
  const int*   idx1    = (const int*)d_in[36];

  // ---- workspace carving (time-aliased regions) ----
  char* ws = (char*)d_ws;
  size_t off = 0;
  auto alloc = [&](size_t bytes) -> char* {
    char* p = ws + off;
    off += (bytes + 255) & ~(size_t)255;
    return p;
  };

  // Region A (33.5MB): X0, later re-used as 4 x 8.4MB bf16 activation slots.
  char* regionA = alloc((size_t)Bb * Nn * 512 * 2);
  __bf16* X0     = (__bf16*)regionA;                        // lives K1 -> G0
  __bf16* xbf    = (__bf16*)regionA;                        // G1 -> G_theta
  __bf16* npo    = (__bf16*)regionA;                        // G_outm -> G5
  __bf16* thetab = (__bf16*)(regionA + 8388608);            // G_theta -> attn
  __bf16* ybf    = (__bf16*)(regionA + 16777216);           // attn -> G_W
  __bf16* zbf    = (__bf16*)(regionA + 25165824);           // G_W -> G_outm
  // Region B (16.8MB): Y0, later h5 + logits.
  char* regionB = alloc((size_t)Bb * Nn * 256 * 2);
  __bf16* Y0     = (__bf16*)regionB;
  __bf16* h5     = (__bf16*)regionB;                        // G5 -> G6
  float*  logits = (float*)(regionB + 2097152);             // G6 -> logsoftmax
  // Persistent f32 features + small buffers.
  float*  xt     = (float*)alloc((size_t)Bb * Nn * CO * 4);
  float*  ipreds = (float*)alloc((size_t)Bb * Nn * CLS * 4);
  float*  diffb  = (float*)alloc((size_t)Bb * Nn * 4);
  int*    Hidx   = (int*)  alloc((size_t)Bb * Hpad * 4);
  __bf16* XH     = (__bf16*)alloc((size_t)Bb * Hpad * 160 * 2);
  __bf16* Hp     = (__bf16*)alloc((size_t)Bb * Hpad * CO * 2);
  __bf16* gxT    = (__bf16*)alloc((size_t)Bb * CO * Hpad * 2);
  __bf16* phib   = (__bf16*)alloc((size_t)Bb * Hpad * CO * 2);
  __bf16* w0bf   = (__bf16*)alloc(256 * 512 * 2);
  __bf16* w1bf   = (__bf16*)alloc(128 * 256 * 2);
  __bf16* whbf   = (__bf16*)alloc(128 * 160 * 2);
  __bf16* woutbf = (__bf16*)alloc(128 * 128 * 2);
  __bf16* wgbf   = (__bf16*)alloc(128 * 128 * 2);
  __bf16* wthbf  = (__bf16*)alloc(128 * 128 * 2);
  __bf16* wphbf  = (__bf16*)alloc(128 * 128 * 2);
  __bf16* wWbf   = (__bf16*)alloc(128 * 128 * 2);
  __bf16* w5bf   = (__bf16*)alloc(32 * 128 * 2);
  __bf16* w6bf   = (__bf16*)alloc(32 * 32 * 2);
  float*  b6p    = (float*)alloc(128);

  auto cvt = [&](const float* s, __bf16* d, int O, int Op, int C, int Cp) {
    int tot = Op * Cp;
    cvt_pad_kernel<<<(tot + 255) / 256, 256, 0, stream>>>(s, d, O, Op, C, Cp);
  };
  cvt(w_mlp0, w0bf, 256, 256, 512, 512);
  cvt(w_mlp1, w1bf, 128, 128, 256, 256);
  cvt(w_h,    whbf, 128, 128, 149, 160);
  cvt(w_outm, woutbf, 128, 128, 128, 128);
  cvt(w_g,    wgbf, 128, 128, 128, 128);
  cvt(w_theta, wthbf, 128, 128, 128, 128);
  cvt(w_phi,  wphbf, 128, 128, 128, 128);
  cvt(w_W,    wWbf, 128, 128, 128, 128);
  cvt(w5,     w5bf, 32, 32, 128, 128);
  cvt(w6,     w6bf, 21, 32, 32, 32);
  pad_bias_kernel<<<1, 32, 0, stream>>>(b6, b6p, CLS, 32);

  // 1) kNN-3 interpolation + concat input
  knn_interp_kernel<<<Bb * (Nn / 256), 256, 0, stream>>>(
      xyz1, xyz2, points1, points2, lastpred, X0, ipreds);

  // 2) MLP0: 512 -> 256, bias+bn+relu
  gemm_bf16<512><<<dim3(Nn / 64, 2, Bb), 128, 0, stream>>>(
      w0bf, 512, X0, (long)Nn * 512, 512, Nn, 256,
      b_mlp0, g_bn0, be_bn0, nullptr, 0, 0, 1,
      Y0, (long)Nn * 256, 256, 0, nullptr, 0, 0, 0);

  // 3) MLP1: 256 -> 128, bias+bn+relu ; dual store f32 xt + bf16 xbf
  gemm_bf16<256><<<dim3(Nn / 64, 1, Bb), 128, 0, stream>>>(
      w1bf, 256, Y0, (long)Nn * 256, 256, Nn, 128,
      b_mlp1, g_bn1, be_bn1, nullptr, 0, 0, 1,
      xbf, (long)Nn * 128, 128, 0, xt, (long)Nn * 128, 128, 0);

  // 4) neighborhood diff + top-k selection
  diff_kernel<<<(Bb * Nn) / 4, 128, 0, stream>>>(xt, idx1, diffb);
  topk_kernel<<<Bb, 1024, 0, stream>>>(diffb, Hidx);
  build_xh_kernel<<<dim3(Hpad, Bb), 192, 0, stream>>>(xt, ipreds, Hidx, XH);

  // 5) Hp = leaky(bn(w_h * XH)) ; 149(pad 160) -> 128
  gemm_bf16<160><<<dim3(Hpad / 64, 1, Bb), 128, 0, stream>>>(
      whbf, 160, XH, (long)Hpad * 160, 160, Hpad, 128,
      nullptr, g_hbn, be_hbn, nullptr, 0, 0, 2,
      Hp, (long)Hpad * 128, 128, 0, nullptr, 0, 0, 0);

  // 6) g_x (stored transposed [c][h]) and phi ([h][c])
  gemm_bf16<128><<<dim3(Hpad / 64, 1, Bb), 128, 0, stream>>>(
      wgbf, 128, Hp, (long)Hpad * 128, 128, Hpad, 128,
      b_g, nullptr, nullptr, nullptr, 0, 0, 0,
      gxT, (long)CO * Hpad, Hpad, 1, nullptr, 0, 0, 0);
  gemm_bf16<128><<<dim3(Hpad / 64, 1, Bb), 128, 0, stream>>>(
      wphbf, 128, Hp, (long)Hpad * 128, 128, Hpad, 128,
      b_phi, nullptr, nullptr, nullptr, 0, 0, 0,
      phib, (long)Hpad * 128, 128, 0, nullptr, 0, 0, 0);

  // 7) theta
  gemm_bf16<128><<<dim3(Nn / 64, 1, Bb), 128, 0, stream>>>(
      wthbf, 128, xbf, (long)Nn * 128, 128, Nn, 128,
      b_theta, nullptr, nullptr, nullptr, 0, 0, 0,
      thetab, (long)Nn * 128, 128, 0, nullptr, 0, 0, 0);

  // 8) softmax(theta . phi^T) @ g_x  (flash, WMMA)
  attn_kernel<<<dim3(Nn / 64, 1, Bb), 128, 0, stream>>>(thetab, phib, gxT, ybf);

  // 9) z = bn(w_W*y + b_W) + x (f32 residual)
  gemm_bf16<128><<<dim3(Nn / 64, 1, Bb), 128, 0, stream>>>(
      wWbf, 128, ybf, (long)Nn * 128, 128, Nn, 128,
      b_W, g_Wbn, be_Wbn, xt, (long)Nn * 128, 128, 0,
      zbf, (long)Nn * 128, 128, 0, nullptr, 0, 0, 0);

  // 10) new_points_out = leaky(bn(w_outm*z)); f32 transposed straight to d_out
  gemm_bf16<128><<<dim3(Nn / 64, 1, Bb), 128, 0, stream>>>(
      woutbf, 128, zbf, (long)Nn * 128, 128, Nn, 128,
      nullptr, g_obn, be_obn, nullptr, 0, 0, 2,
      npo, (long)Nn * 128, 128, 0,
      (float*)d_out, (long)CO * Nn, Nn, 1);

  // 11) h5 = relu(bn(w5*npo + b5)) ; 128 -> 32
  gemm_bf16<128><<<dim3(Nn / 64, 1, Bb), 128, 0, stream>>>(
      w5bf, 128, npo, (long)Nn * 128, 128, Nn, 32,
      b5, g_bn5, be_bn5, nullptr, 0, 0, 1,
      h5, (long)Nn * 32, 32, 0, nullptr, 0, 0, 0);

  // 12) logits = w6*h5 + b6 (padded 21->32)
  gemm_bf16<32><<<dim3(Nn / 64, 1, Bb), 128, 0, stream>>>(
      w6bf, 32, h5, (long)Nn * 32, 32, Nn, 32,
      b6p, nullptr, nullptr, nullptr, 0, 0, 0,
      nullptr, 0, 0, 0, logits, (long)Nn * 32, 32, 0);

  // 13) log_softmax -> d_out tail (B, N, CLS)
  float* pred = (float*)d_out + (size_t)Bb * CO * Nn;
  logsoftmax_kernel<<<(Bb * Nn) / 256, 256, 0, stream>>>(logits, pred);
}
